// GenerateXQs_39754217292543
// MI455X (gfx1250) — compile-verified, matrix operand order checked by reading, not run on hardware
//
#include <hip/hip_runtime.h>
#include <hip/hip_bf16.h>
#include <math.h>

// Problem constants (match setup_inputs)
#define Bq 32
#define Sq 512
#define Hq 1024
#define Tq 8
#define Nq 16

typedef __attribute__((ext_vector_type(16))) __bf16 v16bf;
typedef __attribute__((ext_vector_type(8)))  float  v8f;

// Load one 16-element bf16 fragment: two contiguous 16B runs at p and p+16 elements
// (matches ISA 16-bit A/B layout: K = ko..ko+7 and ko+16..ko+23 per lane).
__device__ __forceinline__ v16bf load_frag(const __bf16* __restrict__ p) {
  v16bf r;
  ((uint4*)&r)[0] = *(const uint4*)p;
  ((uint4*)&r)[1] = *(const uint4*)(p + 16);
  return r;
}

// One-shot f32 -> (bf16 hi, bf16 residual lo) split, strided rows -> packed M*K planes.
__global__ __launch_bounds__(256)
void split_bf16_kernel(const float* __restrict__ X, long row_stride, int K,
                       __bf16* __restrict__ Hi, __bf16* __restrict__ Lo, long total) {
  for (long i = (long)blockIdx.x * 256 + threadIdx.x; i < total; i += (long)gridDim.x * 256) {
    long row = i / K;
    int  col = (int)(i % K);
    float v = X[row * row_stride + col];
    __bf16 h = (__bf16)v;
    Hi[i] = h;
    Lo[i] = (__bf16)(v - (float)h);
  }
}

// C[M,N] = A[M,K] @ W[N,K]^T + bias (+bias2), bf16x3 from pre-split planes, f32 accum.
// Block: 256 threads = 8 waves arranged 2 (M) x 4 (N); block tile 32 x 256.
__global__ __launch_bounds__(256)
void gemm_bf16x3_kernel(const __bf16* __restrict__ Ahi, const __bf16* __restrict__ Alo,
                        const __bf16* __restrict__ Whi, const __bf16* __restrict__ Wlo,
                        const float* __restrict__ bias, const float* __restrict__ bias2,
                        float* __restrict__ C, int ldc, int M, int N, int K)
{
  const int lane = threadIdx.x & 31;
  const int wave = threadIdx.x >> 5;
  const int half = lane >> 4;   // K-group select per ISA 16-bit layout
  const int idx  = lane & 15;   // row (A) / col (B) within tile
  const int i0 = blockIdx.y * 32 + (wave >> 2) * 16;
  const int j0 = blockIdx.x * 256 + (wave & 3) * 64;
  if (i0 >= M) return;          // wave-uniform: EXEC stays all-ones for WMMA

  v8f acc[4] = {v8f{}, v8f{}, v8f{}, v8f{}};
  const __bf16* arow_hi = Ahi + (size_t)(i0 + idx) * (size_t)K;
  const __bf16* arow_lo = Alo + (size_t)(i0 + idx) * (size_t)K;

  for (int kc = 0; kc < K; kc += 32) {
    const int ko = kc + half * 8;
    v16bf a_hi = load_frag(arow_hi + ko);
    v16bf a_lo = load_frag(arow_lo + ko);
    __builtin_prefetch(arow_hi + ko + 32, 0, 1);   // global_prefetch_b8, next K chunk
#pragma unroll
    for (int jt = 0; jt < 4; ++jt) {
      const size_t wro = (size_t)(j0 + jt * 16 + idx) * (size_t)K + ko;
      v16bf b_hi = load_frag(Whi + wro);
      v16bf b_lo = load_frag(Wlo + wro);
      v8f c = acc[jt];
      c = __builtin_amdgcn_wmma_f32_16x16x32_bf16(false, a_hi, false, b_hi, (short)0, c, false, false);
      c = __builtin_amdgcn_wmma_f32_16x16x32_bf16(false, a_lo, false, b_hi, (short)0, c, false, false);
      c = __builtin_amdgcn_wmma_f32_16x16x32_bf16(false, a_hi, false, b_lo, (short)0, c, false, false);
      acc[jt] = c;
    }
  }

#pragma unroll
  for (int jt = 0; jt < 4; ++jt) {
    int col = j0 + jt * 16 + idx;
    float bb = bias ? bias[col] : 0.0f;
    if (bias2) bb += bias2[col];
#pragma unroll
    for (int r = 0; r < 8; ++r) {
      int row = i0 + half * 8 + r;     // C layout: lanes 0-15 M=r, lanes 16-31 M=8+r
      if (row < M) C[(size_t)row * (size_t)ldc + col] = acc[jt][r] + bb;
    }
  }
}

// One softmax-attention: out = softmax(q . keys_s) @ values, over S=512 keys.
// Keys/Vals stored (S,B,H): row (s,b) at (s*B+b)*H. Block = 512 threads (one per s).
__global__ __launch_bounds__(512)
void attend_kernel(const float* __restrict__ Q, long q_bs, long q_ts,
                   const float* __restrict__ Keys, const float* __restrict__ Vals,
                   float* __restrict__ Out, long o_bs, long o_ts,
                   int per_batch)
{
  __shared__ float sh_q[Hq];
  __shared__ float sh_a[Sq];
  __shared__ float sh_red[20];
  const int b = blockIdx.x / per_batch;
  const int t = blockIdx.x % per_batch;
  const float* q = Q + (size_t)b * q_bs + (size_t)t * q_ts;
  float* out = Out + (size_t)b * o_bs + (size_t)t * o_ts;
  const int tid = threadIdx.x;

  for (int h = tid; h < Hq; h += 512) sh_q[h] = q[h];
  __syncthreads();

  // scores: each thread owns one s, streams its own contiguous 4KB key row
  {
    const float* kr = Keys + ((size_t)tid * Bq + b) * Hq;
    float acc = 0.0f;
    for (int h = 0; h < Hq; h += 4) {
      float4 kv = *(const float4*)(kr + h);
      acc = fmaf(sh_q[h],     kv.x, acc);
      acc = fmaf(sh_q[h + 1], kv.y, acc);
      acc = fmaf(sh_q[h + 2], kv.z, acc);
      acc = fmaf(sh_q[h + 3], kv.w, acc);
    }
    sh_a[tid] = acc;
  }
  __syncthreads();

  // softmax over 512 scores: wave32 shfl reduce -> 16 partials -> scalar combine
  float v = sh_a[tid];
  float wm = v;
  for (int off = 16; off > 0; off >>= 1) wm = fmaxf(wm, __shfl_xor(wm, off, 32));
  if ((tid & 31) == 0) sh_red[tid >> 5] = wm;
  __syncthreads();
  if (tid == 0) {
    float m = sh_red[0];
    for (int i = 1; i < 16; ++i) m = fmaxf(m, sh_red[i]);
    sh_red[16] = m;
  }
  __syncthreads();
  float e = __expf(v - sh_red[16]);
  float wsum = e;
  for (int off = 16; off > 0; off >>= 1) wsum += __shfl_xor(wsum, off, 32);
  if ((tid & 31) == 0) sh_red[tid >> 5] = wsum;
  __syncthreads();
  if (tid == 0) {
    float s = 0.0f;
    for (int i = 0; i < 16; ++i) s += sh_red[i];
    sh_red[17] = 1.0f / s;
  }
  __syncthreads();
  sh_a[tid] = e * sh_red[17];
  __syncthreads();

  // weighted value sum, coalesced over h
  for (int h = tid; h < Hq; h += 512) {
    float acc = 0.0f;
    for (int s = 0; s < Sq; ++s)
      acc = fmaf(sh_a[s], Vals[((size_t)s * Bq + b) * Hq + h], acc);
    out[h] = acc;
  }
}

__global__ void copyq_kernel(const float* __restrict__ pq, float* __restrict__ fx) {
  int i = blockIdx.x * 256 + threadIdx.x;
  if (i < Bq * Hq) {
    int b = i / Hq, h = i % Hq;
    fx[(size_t)b * Tq * Hq + h] = pq[i];
  }
}

// outs[0] of the LSTM scan: h0=c0=0 => c = sigmoid(i)*tanh(g), h = sigmoid(o)*tanh(c)
__global__ void lstm_finish_kernel(const float* __restrict__ gates, float* __restrict__ out_qs) {
  int i = blockIdx.x * 256 + threadIdx.x;
  if (i >= Bq * Hq) return;
  int b = i / Hq, h = i % Hq;
  const float* g = gates + (size_t)b * 4 * Hq;
  float ig = 1.0f / (1.0f + __expf(-g[h]));
  float gg = tanhf(g[2 * Hq + h]);
  float og = 1.0f / (1.0f + __expf(-g[3 * Hq + h]));
  float c  = ig * gg;
  out_qs[((size_t)b * (Tq + 1) + Tq) * Hq + h] = og * tanhf(c);
}

__global__ void build_hidden_kernel(const float* __restrict__ hidden,
                                    const float* __restrict__ fx,
                                    float* __restrict__ out) {
  const size_t total = (size_t)(Sq + Tq) * Bq * Hq;
  for (size_t i = (size_t)blockIdx.x * blockDim.x + threadIdx.x; i < total;
       i += (size_t)gridDim.x * blockDim.x) {
    size_t s = i / ((size_t)Bq * Hq);
    size_t rem = i % ((size_t)Bq * Hq);
    if (s < Sq) {
      out[i] = hidden[i];                       // identical (S,B,H) layout
    } else {
      size_t b = rem / Hq, h = rem % Hq;
      out[i] = fx[(b * Tq + (s - Sq)) * Hq + h];
    }
  }
}

__global__ void build_nums_kernel(const float* __restrict__ nums,
                                  const float* __restrict__ fx,
                                  float* __restrict__ out) {
  const int per_b = (Nq + Tq) * Hq;
  int i = blockIdx.x * 256 + threadIdx.x;
  if (i >= Bq * per_b) return;
  int b = i / per_b;
  int rem = i % per_b;
  int j = rem / Hq, h = rem % Hq;
  out[i] = (j < Nq) ? nums[((size_t)b * Nq + j) * Hq + h]
                    : fx[((size_t)b * Tq + (j - Nq)) * Hq + h];
}

extern "C" void kernel_launch(void* const* d_in, const int* in_sizes, int n_in,
                              void* d_out, int out_size, void* d_ws, size_t ws_size,
                              hipStream_t stream) {
  (void)in_sizes; (void)n_in; (void)out_size; (void)ws_size;
  const float* hidden = (const float*)d_in[0];
  const float* nums   = (const float*)d_in[1];
  const float* pq     = (const float*)d_in[2];
  const float* Wk     = (const float*)d_in[3];
  const float* bk     = (const float*)d_in[4];
  const float* Wv     = (const float*)d_in[5];
  const float* bv     = (const float*)d_in[6];
  const float* Wk2    = (const float*)d_in[7];
  const float* bk2    = (const float*)d_in[8];
  const float* Wv2    = (const float*)d_in[9];
  const float* bv2    = (const float*)d_in[10];
  const float* Wih    = (const float*)d_in[11];
  const float* bih    = (const float*)d_in[13];
  const float* bhh    = (const float*)d_in[14];
  // num_xs (d_in[15]) is fixed at T=8 by setup_inputs.

  const size_t SBH = (size_t)Sq * Bq * Hq;     // 16 Mi elements
  const size_t HH  = (size_t)Hq * Hq;          // 1 Mi elements
  char*  base = (char*)d_ws;
  size_t off  = 0;
  auto allocF = [&](size_t n) { float* p = (float*)(base + off); off += n * sizeof(float); return p; };
  auto allocB = [&](size_t n) { __bf16* p = (__bf16*)(base + off); off += n * sizeof(__bf16); return p; };

  float* keys    = allocF(SBH);                // (S,B,H)
  float* values  = allocF(SBH);
  float* keys2   = allocF(SBH);
  float* values2 = allocF(SBH);
  float* fx      = allocF((size_t)Bq * Tq * Hq);   // final_xs (B,T,H)
  float* gates   = allocF((size_t)Bq * 4 * Hq);    // (B,4H)
  __bf16* Ahi    = allocB(SBH);                // hidden split
  __bf16* Alo    = allocB(SBH);
  __bf16* Wkhi   = allocB(HH);   __bf16* Wklo  = allocB(HH);
  __bf16* Wvhi   = allocB(HH);   __bf16* Wvlo  = allocB(HH);
  __bf16* Wk2hi  = allocB(HH);   __bf16* Wk2lo = allocB(HH);
  __bf16* Wv2hi  = allocB(HH);   __bf16* Wv2lo = allocB(HH);
  __bf16* Wihhi  = allocB(4 * HH); __bf16* Wihlo = allocB(4 * HH);
  __bf16* Qshi   = allocB((size_t)Bq * Hq);    // qs[:,0,:] split
  __bf16* Qslo   = allocB((size_t)Bq * Hq);

  float* outp       = (float*)d_out;
  float* out_qs     = outp;                                      // (B, T+1, H)
  float* out_hidden = outp + (size_t)Bq * (Tq + 1) * Hq;         // (S+T, B, H)
  float* out_nums   = out_hidden + (size_t)(Sq + Tq) * Bq * Hq;  // (B, N+T, H)

  // --- one-shot bf16x3 splits (A re-used by 4 GEMMs, W's reused by all M blocks)
  split_bf16_kernel<<<8192, 256, 0, stream>>>(hidden, Hq, Hq, Ahi, Alo, (long)SBH);
  split_bf16_kernel<<<1024, 256, 0, stream>>>(Wk,  Hq, Hq, Wkhi,  Wklo,  (long)HH);
  split_bf16_kernel<<<1024, 256, 0, stream>>>(Wv,  Hq, Hq, Wvhi,  Wvlo,  (long)HH);
  split_bf16_kernel<<<1024, 256, 0, stream>>>(Wk2, Hq, Hq, Wk2hi, Wk2lo, (long)HH);
  split_bf16_kernel<<<1024, 256, 0, stream>>>(Wv2, Hq, Hq, Wv2hi, Wv2lo, (long)HH);
  split_bf16_kernel<<<2048, 256, 0, stream>>>(Wih, Hq, Hq, Wihhi, Wihlo, (long)(4 * HH));

  // --- 4 projection GEMMs: M=S*B=16384, N=K=1024 (pure WMMA inner loop)
  dim3 tb(256);
  dim3 gproj(Hq / 256, (Sq * Bq) / 32);
  gemm_bf16x3_kernel<<<gproj, tb, 0, stream>>>(Ahi, Alo, Wkhi,  Wklo,  bk,  nullptr, keys,    Hq, Sq * Bq, Hq, Hq);
  gemm_bf16x3_kernel<<<gproj, tb, 0, stream>>>(Ahi, Alo, Wvhi,  Wvlo,  bv,  nullptr, values,  Hq, Sq * Bq, Hq, Hq);
  gemm_bf16x3_kernel<<<gproj, tb, 0, stream>>>(Ahi, Alo, Wk2hi, Wk2lo, bk2, nullptr, keys2,   Hq, Sq * Bq, Hq, Hq);
  gemm_bf16x3_kernel<<<gproj, tb, 0, stream>>>(Ahi, Alo, Wv2hi, Wv2lo, bv2, nullptr, values2, Hq, Sq * Bq, Hq, Hq);

  // final_xs[:,0,:] = problem_q
  copyq_kernel<<<(Bq * Hq + 255) / 256, 256, 0, stream>>>(pq, fx);

  // serial generation: x_t = attend(x_{t-1}, keys, values)
  for (int t = 1; t < Tq; ++t) {
    attend_kernel<<<Bq, 512, 0, stream>>>(fx + (size_t)(t - 1) * Hq, (long)Tq * Hq, 0L,
                                          keys, values,
                                          fx + (size_t)t * Hq, (long)Tq * Hq, 0L, 1);
  }

  // second attention: qs -> directly into final_qs rows 0..T-1
  attend_kernel<<<Bq * Tq, 512, 0, stream>>>(fx, (long)Tq * Hq, (long)Hq,
                                             keys2, values2,
                                             out_qs, (long)(Tq + 1) * Hq, (long)Hq, Tq);

  // LSTM collapses to one step: gates = qs[:,0,:] @ W_ih^T + b_ih + b_hh  (h0 = 0)
  split_bf16_kernel<<<128, 256, 0, stream>>>(out_qs, (long)(Tq + 1) * Hq, Hq,
                                             Qshi, Qslo, (long)Bq * Hq);
  dim3 ggate((4 * Hq) / 256, 1);
  gemm_bf16x3_kernel<<<ggate, tb, 0, stream>>>(Qshi, Qslo, Wihhi, Wihlo, bih, bhh,
                                               gates, 4 * Hq, Bq, 4 * Hq, Hq);
  lstm_finish_kernel<<<(Bq * Hq + 255) / 256, 256, 0, stream>>>(gates, out_qs);

  // assemble concatenated outputs
  build_hidden_kernel<<<4096, 256, 0, stream>>>(hidden, fx, out_hidden);
  build_nums_kernel<<<(Bq * (Nq + Tq) * Hq + 255) / 256, 256, 0, stream>>>(nums, fx, out_nums);
}
